// Linear_extractor_cluster_1142461300768
// MI455X (gfx1250) — matrix operand order repaired and checked by reading, not compiled, not run on hardware
//
#include <hip/hip_runtime.h>

typedef __attribute__((ext_vector_type(16))) __bf16 v16bf;
typedef __attribute__((ext_vector_type(8)))  float  v8f;

#define N_TOK 16384
#define DDIM  512
#define HDIM  1024
#define NEXP  8
#define CAP   5120
#define MT    64                 // rows per expert tile
#define TILES_PER_E (CAP / MT)   // 80

__device__ __forceinline__ __bf16 f2bf(float f) {
    union { float f; unsigned u; } v; v.f = f;
    unsigned r = v.u + 0x7FFFu + ((v.u >> 16) & 1u);   // round-to-nearest-even
    unsigned short hs = (unsigned short)(r >> 16);
    __bf16 b;
    __builtin_memcpy(&b, &hs, 2);
    return b;
}

// ---------------------------------------------------------------- zero output
__global__ void zero_out_kernel(float4* __restrict__ out) {
    size_t i = (size_t)blockIdx.x * blockDim.x + threadIdx.x;   // N*D/4 elements
    out[i] = make_float4(0.f, 0.f, 0.f, 0.f);
}

// ---------------------------------------------------------------- x -> bf16 (row-major)
__global__ __launch_bounds__(256)
void convert_x_kernel(const float* __restrict__ x, __bf16* __restrict__ xb) {
    size_t j = (size_t)blockIdx.x * blockDim.x + threadIdx.x;   // N*D/8 threads
    const float4* s = (const float4*)(x + j * 8);
    float4 a = s[0], b = s[1];
    union { unsigned short us[8]; uint4 u; } p;
    __bf16 t0 = f2bf(a.x), t1 = f2bf(a.y), t2 = f2bf(a.z), t3 = f2bf(a.w);
    __bf16 t4 = f2bf(b.x), t5 = f2bf(b.y), t6 = f2bf(b.z), t7 = f2bf(b.w);
    __builtin_memcpy(&p.us[0], &t0, 2); __builtin_memcpy(&p.us[1], &t1, 2);
    __builtin_memcpy(&p.us[2], &t2, 2); __builtin_memcpy(&p.us[3], &t3, 2);
    __builtin_memcpy(&p.us[4], &t4, 2); __builtin_memcpy(&p.us[5], &t5, 2);
    __builtin_memcpy(&p.us[6], &t6, 2); __builtin_memcpy(&p.us[7], &t7, 2);
    *(uint4*)(xb + j * 8) = p.u;
}

// ---------------------------------------------------------------- gating + top-2
__global__ __launch_bounds__(256)
void gate_topk_kernel(const float* __restrict__ x, const float* __restrict__ eps,
                      const float* __restrict__ wg, const float* __restrict__ wn,
                      const float* __restrict__ Wh,
                      int* __restrict__ topidx, float* __restrict__ topgate) {
    int n = blockIdx.x * blockDim.x + threadIdx.x;
    if (n >= N_TOK) return;
    float accg[NEXP], accn[NEXP];
#pragma unroll
    for (int e = 0; e < NEXP; ++e) { accg[e] = 0.f; accn[e] = 0.f; }

    const float4* x4  = (const float4*)(x + (size_t)n * DDIM);
    const float4* wg4 = (const float4*)wg;
    const float4* wn4 = (const float4*)wn;
    for (int d4 = 0; d4 < DDIM / 4; ++d4) {
        float4 xv = x4[d4];
        float xs[4] = { xv.x, xv.y, xv.z, xv.w };
#pragma unroll
        for (int u = 0; u < 4; ++u) {
            int base = d4 * 8 + u * 2;
            float4 ga = wg4[base], gb = wg4[base + 1];
            float4 na = wn4[base], nb = wn4[base + 1];
            float s = xs[u];
            accg[0] += s * ga.x; accg[1] += s * ga.y; accg[2] += s * ga.z; accg[3] += s * ga.w;
            accg[4] += s * gb.x; accg[5] += s * gb.y; accg[6] += s * gb.z; accg[7] += s * gb.w;
            accn[0] += s * na.x; accn[1] += s * na.y; accn[2] += s * na.z; accn[3] += s * na.w;
            accn[4] += s * nb.x; accn[5] += s * nb.y; accn[6] += s * nb.z; accn[7] += s * nb.w;
        }
    }
    float nl[NEXP];
#pragma unroll
    for (int e = 0; e < NEXP; ++e) {
        float c = 0.f;
#pragma unroll
        for (int j = 0; j < NEXP; ++j) c += accg[j] * Wh[j * NEXP + e];
        float v = accn[e];
        float sp = (v > 20.f) ? v : log1pf(expf(v));
        nl[e] = c + eps[(size_t)n * NEXP + e] * (sp + 1e-2f);
    }
    int i0 = 0, i1 = -1; float v0 = nl[0], v1 = -3.4e38f;
#pragma unroll
    for (int e = 1; e < NEXP; ++e) {
        float v = nl[e];
        if (v > v0) { v1 = v0; i1 = i0; v0 = v; i0 = e; }
        else if (v > v1) { v1 = v; i1 = e; }
    }
    float e1 = expf(v1 - v0);
    float g0 = 1.f / (1.f + e1);
    topidx[n * 2 + 0] = i0;  topidx[n * 2 + 1] = i1;
    topgate[n * 2 + 0] = g0; topgate[n * 2 + 1] = 1.f - g0;
}

// ---------------------------------------------------------------- per-block expert histogram
__global__ void hist_kernel(const int* __restrict__ topidx, int* __restrict__ partial) {
    __shared__ int h[NEXP];
    int tid = threadIdx.x;
    if (tid < NEXP) h[tid] = 0;
    __syncthreads();
    int n = blockIdx.x * 256 + tid;
    atomicAdd(&h[topidx[n * 2 + 0]], 1);
    atomicAdd(&h[topidx[n * 2 + 1]], 1);
    __syncthreads();
    if (tid < NEXP) partial[blockIdx.x * NEXP + tid] = h[tid];
}

// ---------------------------------------------------------------- exclusive scan of block counts
__global__ void scan_kernel(const int* __restrict__ partial, int* __restrict__ base,
                            int* __restrict__ totals) {
    int e = threadIdx.x;
    if (e >= NEXP) return;
    int run = 0;
    for (int b = 0; b < N_TOK / 256; ++b) {
        base[b * NEXP + e] = run;
        run += partial[b * NEXP + e];
    }
    totals[e] = run;
}

// ---------------------------------------------------------------- in-order slot assignment
__global__ void dispatch_kernel(const int* __restrict__ topidx, const float* __restrict__ topgate,
                                const int* __restrict__ base,
                                int* __restrict__ buf, float* __restrict__ gbuf) {
    int e = threadIdx.x;
    if (e >= NEXP) return;
    int b = blockIdx.x;
    int cnt = base[b * NEXP + e];
    for (int t = 0; t < 256; ++t) {
        int n = b * 256 + t;
#pragma unroll
        for (int j = 0; j < 2; ++j) {
            if (topidx[n * 2 + j] == e) {
                if (cnt < CAP) {
                    buf[e * CAP + cnt]  = n;
                    gbuf[e * CAP + cnt] = topgate[n * 2 + j];
                }
                ++cnt;
            }
        }
    }
}

// ---------------------------------------------------------------- fill unused slots
__global__ void fill_kernel(const int* __restrict__ totals,
                            int* __restrict__ buf, float* __restrict__ gbuf) {
    int i = blockIdx.x * blockDim.x + threadIdx.x;       // E*CAP threads
    int e = i / CAP, c = i % CAP;
    int filled = totals[e]; if (filled > CAP) filled = CAP;
    if (c >= filled) { buf[i] = 0; gbuf[i] = 0.f; }
}

// ---------------------------------------------------------------- repack W1 -> bf16 B-fragment layout
// W1b layout: [e][kc(16)][hTile(64)][lane(32)][e2(16)],  K = kc*32 + (lane>>4)*8 + e2 + (e2&8),
//             col = hTile*16 + (lane&15)
__global__ __launch_bounds__(256)
void convert_w1_kernel(const float* __restrict__ W1, __bf16* __restrict__ W1b) {
    int i = blockIdx.x * blockDim.x + threadIdx.x;       // 8*16*64*32 threads
    int lane  = i & 31;
    int hTile = (i >> 5) & 63;
    int kc    = (i >> 11) & 15;
    int e     = i >> 15;
    int khalf = lane >> 4, n = lane & 15;
    const float* src = W1 + ((size_t)e * DDIM + kc * 32 + khalf * 8) * HDIM + hTile * 16 + n;
    v16bf frag;
#pragma unroll
    for (int e2 = 0; e2 < 16; ++e2) frag[e2] = f2bf(src[(size_t)(e2 + (e2 & 8)) * HDIM]);
    *(v16bf*)(W1b + (size_t)i * 16) = frag;
}

// ---------------------------------------------------------------- repack W2 -> bf16 B-fragment layout
// W2b layout: [e][hc(32)][dTile(32)][lane(32)][e2(16)],  K = hc*32 + (lane>>4)*8 + e2 + (e2&8),
//             col = dTile*16 + (lane&15)
__global__ __launch_bounds__(256)
void convert_w2_kernel(const float* __restrict__ W2, __bf16* __restrict__ W2b) {
    int i = blockIdx.x * blockDim.x + threadIdx.x;       // 8*32*32*32 threads
    int lane  = i & 31;
    int dTile = (i >> 5) & 31;
    int hc    = (i >> 10) & 31;
    int e     = i >> 15;
    int khalf = lane >> 4, n = lane & 15;
    const float* src = W2 + ((size_t)e * HDIM + hc * 32 + khalf * 8) * DDIM + dTile * 16 + n;
    v16bf frag;
#pragma unroll
    for (int e2 = 0; e2 < 16; ++e2) frag[e2] = f2bf(src[(size_t)(e2 + (e2 & 8)) * DDIM]);
    *(v16bf*)(W2b + (size_t)i * 16) = frag;
}

// ---------------------------------------------------------------- fused expert FFN (WMMA bf16)
__global__ __launch_bounds__(256)
void moe_expert_kernel(const __bf16* __restrict__ xb,
                       const __bf16* __restrict__ W1b, const float* __restrict__ b1,
                       const __bf16* __restrict__ W2b, const float* __restrict__ b2,
                       const int* __restrict__ buf, const float* __restrict__ gbuf,
                       float* __restrict__ out) {
    extern __shared__ __align__(32) char smem_raw[];
    __bf16* As = (__bf16*)smem_raw;              // [4 mt][16 kc][32 lane][16 e] = 64 KB
    __bf16* Hs = (__bf16*)(smem_raw + 65536);    // [4 mt][32 lane][16 e]       = 4 KB

    int e    = blockIdx.x / TILES_PER_E;
    int row0 = (blockIdx.x % TILES_PER_E) * MT;
    int tid  = threadIdx.x;
    int lane = tid & 31, wid = tid >> 5;

    const __bf16* W1be = W1b + (size_t)e * DDIM * HDIM;   // fragment-packed
    const __bf16* W2be = W2b + (size_t)e * HDIM * DDIM;
    const int*    bufe  = buf  + e * CAP + row0;
    const float*  gbufe = gbuf + e * CAP + row0;

    // ---- stage gathered bf16 token rows into LDS in WMMA A-fragment layout ----
    // 8 consecutive k (k%8==0) stay inside one fragment 8-group: 16B load -> 16B LDS store
    for (int i = tid; i < MT * (DDIM / 8); i += 256) {
        int row = i >> 6, c8 = i & 63;
        int token = bufe[row];
        uint4 v = *(const uint4*)(xb + (size_t)token * DDIM + c8 * 8);
        int k  = c8 << 3;
        int mt = row >> 4, mm = row & 15;
        int kc = k >> 5,  kk = k & 31;
        int half = (kk >> 3) & 1;
        int eb   = (kk & 16) >> 1;                  // 0 or 8
        *(uint4*)(As + (size_t)(((((mt << 4) + kc) << 5) + mm + (half << 4)) * 16 + eb)) = v;
    }
    __syncthreads();

    int mtw = wid & 3;            // m-strip (16 rows) this wave owns
    int nh  = wid >> 2;           // 0..1
    int nlo = lane & 15, khalf = lane >> 4;

    v8f yacc[16];
    v8f zero = {};
#pragma unroll
    for (int t = 0; t < 16; ++t) yacc[t] = zero;

    for (int hc = 0; hc < HDIM / 32; ++hc) {
        int hTile = hc * 2 + nh;
        int hcol  = hTile * 16 + nlo;

        // prefetch next chunk's B fragments while this chunk computes
        if (hc + 1 < HDIM / 32) {
            __builtin_prefetch(W1be + (size_t)((hTile + 2) * 32 + lane) * 16, 0, 1);
            __builtin_prefetch(W2be + (size_t)((((hc + 1) << 5) + (nh << 4)) * 32 + lane) * 16, 0, 1);
        }

        // ---- GEMM1: h tile [16x16], K swept 16 x 32 ----
        v8f hacc = zero;
#pragma unroll 4
        for (int kc = 0; kc < 16; ++kc) {
            v16bf a = *(const v16bf*)(As + (size_t)(((((mtw << 4) + kc) << 5) + lane) * 16));
            v16bf b = *(const v16bf*)(W1be + (size_t)(((kc << 6) + hTile) * 32 + lane) * 16);
            hacc = __builtin_amdgcn_wmma_f32_16x16x32_bf16(false, a, false, b,
                                                           (short)0, hacc, false, false);
        }
        // bias + ReLU, restage h chunk into LDS as GEMM2's A operand
        float bias1 = b1[e * HDIM + hcol];
        int kkH  = (nh << 4) + nlo;
        int hafH = (kkH >> 3) & 1;
        int eH   = (kkH & 7) + ((kkH & 16) >> 1);
#pragma unroll
        for (int r = 0; r < 8; ++r) {
            float hv = hacc[r] + bias1;
            hv = hv > 0.f ? hv : 0.f;
            int mm = r + (khalf << 3);
            Hs[(size_t)(((mtw << 5) + mm + (hafH << 4)) * 16 + eH)] = f2bf(hv);
        }
        __syncthreads();

        // ---- GEMM2: y[16 x 256] += h_chunk[16x32] @ W2[32, 256cols] ----
        v16bf a2 = *(const v16bf*)(Hs + (size_t)(((mtw << 5) + lane) * 16));
        const __bf16* w2base = W2be + (size_t)(((hc << 5) + (nh << 4)) * 32 + lane) * 16;
#pragma unroll 4
        for (int t = 0; t < 16; ++t) {
            v16bf b = *(const v16bf*)(w2base + (size_t)(t * 32) * 16);
            yacc[t] = __builtin_amdgcn_wmma_f32_16x16x32_bf16(false, a2, false, b,
                                                              (short)0, yacc[t], false, false);
        }
        __syncthreads();
    }

    // ---- epilogue: + b2, * gate, scatter-add to tokens (skip gate==0 slots) ----
    float gv[8]; int tokv[8];
#pragma unroll
    for (int r = 0; r < 8; ++r) {
        int m = (mtw << 4) + r + (khalf << 3);
        gv[r]   = gbufe[m];
        tokv[r] = bufe[m];
    }
#pragma unroll
    for (int t = 0; t < 16; ++t) {
        int col = (nh << 8) + (t << 4) + nlo;
        float bias2 = b2[e * DDIM + col];
#pragma unroll
        for (int r = 0; r < 8; ++r) {
            if (gv[r] != 0.f) {
                float yv = (yacc[t][r] + bias2) * gv[r];
                atomicAdd(out + (size_t)tokv[r] * DDIM + col, yv);
            }
        }
    }
}

// ---------------------------------------------------------------- launcher
extern "C" void kernel_launch(void* const* d_in, const int* in_sizes, int n_in,
                              void* d_out, int out_size, void* d_ws, size_t ws_size,
                              hipStream_t stream) {
    const float* x   = (const float*)d_in[0];
    const float* eps = (const float*)d_in[1];
    const float* wg  = (const float*)d_in[2];
    const float* wn  = (const float*)d_in[3];
    const float* Wh  = (const float*)d_in[4];
    const float* W1  = (const float*)d_in[5];
    const float* b1  = (const float*)d_in[6];
    const float* W2  = (const float*)d_in[7];
    const float* b2  = (const float*)d_in[8];
    float* out = (float*)d_out;

    char* ws = (char*)d_ws;
    int*    topidx  = (int*)   (ws);                     // 131072 B
    float*  topgate = (float*) (ws + 131072);            // 131072 B
    int*    partial = (int*)   (ws + 262144);            //   2048 B
    int*    base    = (int*)   (ws + 264192);            //   2048 B
    int*    totals  = (int*)   (ws + 266240);            //     32 B
    int*    buf     = (int*)   (ws + 266272);            // 163840 B
    float*  gbuf    = (float*) (ws + 430112);            // 163840 B
    __bf16* W1b     = (__bf16*)(ws + 593952);            // 8 MB
    __bf16* W2b     = (__bf16*)(ws + 593952 + 8388608);  // 8 MB
    __bf16* xb      = (__bf16*)(ws + 593952 + 16777216); // 16 MB

    zero_out_kernel<<<(N_TOK * DDIM / 4) / 256, 256, 0, stream>>>((float4*)out);
    convert_x_kernel<<<(N_TOK * DDIM / 8) / 256, 256, 0, stream>>>(x, xb);
    convert_w1_kernel<<<(NEXP * DDIM * HDIM / 16) / 256, 256, 0, stream>>>(W1, W1b);
    convert_w2_kernel<<<(NEXP * HDIM * DDIM / 16) / 256, 256, 0, stream>>>(W2, W2b);
    gate_topk_kernel<<<N_TOK / 256, 256, 0, stream>>>(x, eps, wg, wn, Wh, topidx, topgate);
    hist_kernel<<<N_TOK / 256, 256, 0, stream>>>(topidx, partial);
    scan_kernel<<<1, 32, 0, stream>>>(partial, base, totals);
    dispatch_kernel<<<N_TOK / 256, 32, 0, stream>>>(topidx, topgate, base, buf, gbuf);
    fill_kernel<<<(NEXP * CAP) / 256, 256, 0, stream>>>(totals, buf, gbuf);
    moe_expert_kernel<<<NEXP * TILES_PER_E, 256, 69632, stream>>>(xb, W1b, b1, W2b, b2,
                                                                  buf, gbuf, out);
}
